// ClientModel_18966575579618
// MI455X (gfx1250) — compile-verified, matrix operand order checked by reading, not run on hardware
//
#include <hip/hip_runtime.h>

// ---------------- problem constants ----------------
#define SEQ_LEN     80
#define NUM_CLASSES 80
#define NH          256     // hidden
#define EMB         8
#define BATCH       8192

// ---------------- tiling constants -----------------
#define BT   64             // batch rows per workgroup
#define MT   4              // BT/16 M-tiles
#define K1T  9              // ceil((EMB+NH)/32) -> K padded to 288
#define K2T  16             // (NH+NH)/32
#define NT   64             // 4*NH/16 N-tiles of the gate matmul
#define S1   296            // xh1 LDS row stride (bf16 elems), 16B aligned, bank-friendly
#define S2   520            // xh2 LDS row stride
#define SH   264            // hst (h staging) row stride
#define NGRP 16             // NH/16 n-groups

typedef __bf16 bf16_t;
typedef __attribute__((ext_vector_type(16))) __bf16 v16bf;
typedef __attribute__((ext_vector_type(8)))  float  v8f;
typedef __attribute__((ext_vector_type(4)))  float  f32x4;
typedef __attribute__((ext_vector_type(4)))  unsigned int u32x4;

union FragU { v16bf v; u32x4 q[2]; };

__device__ __forceinline__ bf16_t f2bf(float f) {
  unsigned u = __float_as_uint(f);
  u += 0x7fffu + ((u >> 16) & 1u);           // round-to-nearest-even
  unsigned short s = (unsigned short)(u >> 16);
  return __builtin_bit_cast(bf16_t, s);
}
__device__ __forceinline__ float frcp(float x) {
#if __has_builtin(__builtin_amdgcn_rcpf)
  return __builtin_amdgcn_rcpf(x);
#else
  return 1.0f / x;
#endif
}
__device__ __forceinline__ float sigf(float x) { return frcp(1.0f + __expf(-x)); }
__device__ __forceinline__ float tanhfast(float x) {
#if __has_builtin(__builtin_amdgcn_tanhf)
  return __builtin_amdgcn_tanhf(x);          // gfx1250 v_tanh_f32 if available
#else
  float e = __expf(-2.0f * x);
  return (1.0f - e) * frcp(1.0f + e);
#endif
}

// A fragment (16x32 bf16): lane<16 holds row M=lane&15, K {k0..k0+7, k0+16..k0+23};
// lane>=16 same row, K shifted by +8. Two 16B LDS loads per lane.
__device__ __forceinline__ v16bf load_a_frag(const bf16_t* rowp, int k0, int lane) {
  int kadj = k0 + ((lane >= 16) ? 8 : 0);
  FragU u;
  u.q[0] = *(const u32x4*)(rowp + kadj);
  u.q[1] = *(const u32x4*)(rowp + kadj + 16);
  return u.v;
}

// B fragment: pre-packed 1KB tile, lane-major; lane reads its own 32 contiguous bytes.
__device__ __forceinline__ v16bf load_b_frag(const bf16_t* tb, int lane) {
  FragU u;
  const u32x4* p = (const u32x4*)tb + (size_t)lane * 2;
  u.q[0] = p[0];
  u.q[1] = p[1];
  return u.v;
}

// -------- pack fp32 [Ksrc,1024] weights into WMMA B-fragment-ordered bf16 tiles --------
__global__ __launch_bounds__(256) void pack_weights(const float* __restrict__ W, int Ksrc,
                                                    int total, bf16_t* __restrict__ dst) {
  int id = blockIdx.x * 256 + threadIdx.x;
  if (id >= total) return;
  int tile = id >> 9;            // 512 bf16 per tile
  int lane = (id >> 4) & 31;
  int e    = id & 15;
  int kt = tile >> 6;            // NT = 64
  int nt = tile & 63;
  int n = nt * 16 + (lane & 15);
  int k = kt * 32 + (e & 7) + ((e >= 8) ? 16 : 0) + ((lane >= 16) ? 8 : 0);
  float v = (k < Ksrc) ? W[(size_t)k * (4 * NH) + n] : 0.0f;
  dst[id] = f2bf(v);
}

// -------- GEMM for one 16-column n-group: acc[q] = xh @ W[:, gate q cols] --------
template <int KT, int STRIDE>
__device__ __forceinline__ void gemm_group(const bf16_t* __restrict__ xh,
                                           const bf16_t* __restrict__ Wp,
                                           int ngrp, int lane, v8f acc[4][MT]) {
  const int hlane = lane & 15;
  for (int kt = 0; kt < KT; ++kt) {
    v16bf a[MT];
#pragma unroll
    for (int m = 0; m < MT; ++m)
      a[m] = load_a_frag(xh + (size_t)(m * 16 + hlane) * STRIDE, kt * 32, lane);
#pragma unroll
    for (int q = 0; q < 4; ++q) {
      v16bf bb = load_b_frag(Wp + (size_t)(kt * NT + q * 16 + ngrp) * 512, lane);
#pragma unroll
      for (int m = 0; m < MT; ++m)
        acc[q][m] = __builtin_amdgcn_wmma_f32_16x16x32_bf16(
            false, a[m], false, bb, (short)0, acc[q][m], false, false);
    }
  }
}

// -------- persistent 2-layer LSTM over 80 steps; one WG owns 64 batch rows --------
__global__ __launch_bounds__(256, 1) void lstm_persist(
    const int* __restrict__ features, const float* __restrict__ emb,
    const float* __restrict__ b1, const float* __restrict__ b2,
    const bf16_t* __restrict__ W1p, const bf16_t* __restrict__ W2p,
    float* __restrict__ h2out) {
  __shared__ __align__(16) bf16_t xh1[BT][S1];   // [x(8) | h1(256) | pad]
  __shared__ __align__(16) bf16_t xh2[BT][S2];   // [h1(256) | h2(256) | pad]
  __shared__ __align__(16) bf16_t hst[BT][SH];   // h_new staging
  __shared__ __align__(16) float  cbuf[2][NGRP * MT * 256];  // c-state, accumulator layout
  __shared__ bf16_t embs[NUM_CLASSES * EMB];
  __shared__ float  biaS[2][4 * NH];             // b1, b2

  const int tid  = threadIdx.x;
  const int wave = tid >> 5, lane = tid & 31;
  const int hlane = lane & 15, hsel = lane >> 4;
  const int b0 = blockIdx.x * BT;

  for (int i = tid; i < BT * S1; i += 256) ((unsigned short*)xh1)[i] = 0;
  for (int i = tid; i < BT * S2; i += 256) ((unsigned short*)xh2)[i] = 0;
  for (int i = tid; i < 2 * NGRP * MT * 256; i += 256) ((float*)cbuf)[i] = 0.0f;
  for (int i = tid; i < NUM_CLASSES * EMB; i += 256) embs[i] = f2bf(emb[i]);
  for (int i = tid; i < 4 * NH; i += 256) {
    biaS[0][i] = b1[i];
    biaS[1][i] = b2[i];
  }

  const v8f vz = {0.f, 0.f, 0.f, 0.f, 0.f, 0.f, 0.f, 0.f};
  __syncthreads();

#pragma unroll 1
  for (int t = 0; t < SEQ_LEN; ++t) {
    // embedding lookup for this timestep into xh1[:, 0:8]
    for (int i = tid; i < BT * EMB; i += 256) {
      int bl = i >> 3, e = i & 7;
      int idx = features[(size_t)(b0 + bl) * SEQ_LEN + t];
      xh1[bl][e] = embs[idx * EMB + e];
    }
    __syncthreads();

    // ---- layer 1: [x|h1] @ W1; gates fused; h_new -> staging ----
#pragma unroll 1
    for (int g = 0; g < 2; ++g) {
      const int ngrp = wave * 2 + g;
      const int n = ngrp * 16 + hlane;
      v8f acc[4][MT];
#pragma unroll
      for (int q = 0; q < 4; ++q)
#pragma unroll
        for (int m = 0; m < MT; ++m) acc[q][m] = vz;
      gemm_group<K1T, S1>(&xh1[0][0], W1p, ngrp, lane, acc);
      float bi = biaS[0][0 * NH + n];
      float bj = biaS[0][1 * NH + n];
      float bf = biaS[0][2 * NH + n] + 1.0f;       // FORGET_BIAS folded in
      float bo = biaS[0][3 * NH + n];
#pragma unroll
      for (int m = 0; m < MT; ++m) {
        float* cp = &cbuf[0][(ngrp * MT + m) * 256 + lane * 8];
        f32x4 clo = *(f32x4*)cp;
        f32x4 chi = *(f32x4*)(cp + 4);
#pragma unroll
        for (int r = 0; r < 8; ++r) {
          float iv = acc[0][m][r] + bi;
          float jv = acc[1][m][r] + bj;
          float fv = acc[2][m][r] + bf;
          float ov = acc[3][m][r] + bo;
          float co = (r < 4) ? clo[r & 3] : chi[r & 3];
          float c  = co * sigf(fv) + sigf(iv) * tanhfast(jv);
          if (r < 4) clo[r & 3] = c; else chi[r & 3] = c;
          hst[m * 16 + r + hsel * 8][n] = f2bf(tanhfast(c) * sigf(ov));
        }
        *(f32x4*)cp       = clo;
        *(f32x4*)(cp + 4) = chi;
      }
    }
    __syncthreads();                       // all GEMM reads of h1_old done; hst complete
    // scatter h1_new into operand buffers (16B LDS chunks)
    for (int cch = tid; cch < BT * NH / 8; cch += 256) {
      int row = cch >> 5;                  // 32 chunks of 8 per row
      int col = (cch & 31) * 8;
      u32x4 v = *(const u32x4*)&hst[row][col];
      *(u32x4*)&xh1[row][EMB + col] = v;   // next step's layer-1 input
      *(u32x4*)&xh2[row][col]       = v;   // this step's layer-2 input
    }
    __syncthreads();

    // ---- layer 2: [h1|h2] @ W2 ----
#pragma unroll 1
    for (int g = 0; g < 2; ++g) {
      const int ngrp = wave * 2 + g;
      const int n = ngrp * 16 + hlane;
      v8f acc[4][MT];
#pragma unroll
      for (int q = 0; q < 4; ++q)
#pragma unroll
        for (int m = 0; m < MT; ++m) acc[q][m] = vz;
      gemm_group<K2T, S2>(&xh2[0][0], W2p, ngrp, lane, acc);
      float bi = biaS[1][0 * NH + n];
      float bj = biaS[1][1 * NH + n];
      float bf = biaS[1][2 * NH + n] + 1.0f;
      float bo = biaS[1][3 * NH + n];
#pragma unroll
      for (int m = 0; m < MT; ++m) {
        float* cp = &cbuf[1][(ngrp * MT + m) * 256 + lane * 8];
        f32x4 clo = *(f32x4*)cp;
        f32x4 chi = *(f32x4*)(cp + 4);
#pragma unroll
        for (int r = 0; r < 8; ++r) {
          float iv = acc[0][m][r] + bi;
          float jv = acc[1][m][r] + bj;
          float fv = acc[2][m][r] + bf;
          float ov = acc[3][m][r] + bo;
          float co = (r < 4) ? clo[r & 3] : chi[r & 3];
          float c  = co * sigf(fv) + sigf(iv) * tanhfast(jv);
          if (r < 4) clo[r & 3] = c; else chi[r & 3] = c;
          int row = m * 16 + r + hsel * 8;
          float hv = tanhfast(c) * sigf(ov);
          hst[row][n] = f2bf(hv);
          if (t == SEQ_LEN - 1)
            h2out[(size_t)(b0 + row) * NH + n] = hv;   // f32 for the loss head
        }
        *(f32x4*)cp       = clo;
        *(f32x4*)(cp + 4) = chi;
      }
    }
    __syncthreads();
    for (int cch = tid; cch < BT * NH / 8; cch += 256) {
      int row = cch >> 5;
      int col = (cch & 31) * 8;
      *(u32x4*)&xh2[row][NH + col] = *(const u32x4*)&hst[row][col];
    }
    __syncthreads();
  }
}

// -------- dense + log-softmax loss; one wave32 per batch row --------
__global__ __launch_bounds__(256) void dense_loss(const float* __restrict__ h2,
                                                  const float* __restrict__ Wd,
                                                  const float* __restrict__ bd,
                                                  const int* __restrict__ labels,
                                                  float* __restrict__ out) {
  int row  = blockIdx.x * 8 + (threadIdx.x >> 5);
  int lane = threadIdx.x & 31;
  const float* h = h2 + (size_t)row * NH;
  int c0 = lane, c1i = lane + 32, c2 = lane + 64;
  int c2c = (c2 < NUM_CLASSES) ? c2 : 0;
  float a0 = bd[c0], a1 = bd[c1i], a2 = bd[c2c];
  for (int k = 0; k < NH; ++k) {
    float hk = h[k];
    const float* w = Wd + (size_t)k * NUM_CLASSES;
    a0 += hk * w[c0];
    a1 += hk * w[c1i];
    a2 += hk * w[c2c];
  }
  if (c2 >= NUM_CLASSES) a2 = -1e30f;

  float m = fmaxf(a0, fmaxf(a1, a2));
#pragma unroll
  for (int off = 16; off; off >>= 1) m = fmaxf(m, __shfl_xor(m, off, 32));
  float s = __expf(a0 - m) + __expf(a1 - m) + __expf(a2 - m);
#pragma unroll
  for (int off = 16; off; off >>= 1) s += __shfl_xor(s, off, 32);
  float lse = m + __logf(s);

  int lb = labels[row];
  float sel = (c0 == lb) ? a0 : ((c1i == lb) ? a1 : ((c2 == lb) ? a2 : 0.0f));
#pragma unroll
  for (int off = 16; off; off >>= 1) sel += __shfl_xor(sel, off, 32);
  if (lane == 0) out[row] = lse - sel;
}

extern "C" void kernel_launch(void* const* d_in, const int* in_sizes, int n_in,
                              void* d_out, int out_size, void* d_ws, size_t ws_size,
                              hipStream_t stream) {
  (void)in_sizes; (void)n_in; (void)out_size; (void)ws_size;
  const int*   features  = (const int*)d_in[0];
  const int*   labels    = (const int*)d_in[1];
  const float* embedding = (const float*)d_in[2];
  const float* W1 = (const float*)d_in[3];
  const float* b1 = (const float*)d_in[4];
  const float* W2 = (const float*)d_in[5];
  const float* b2 = (const float*)d_in[6];
  const float* Wd = (const float*)d_in[7];
  const float* bd = (const float*)d_in[8];
  float* out = (float*)d_out;

  char* ws = (char*)d_ws;
  bf16_t* W1p = (bf16_t*)ws;                               // 9*64*1KB  = 576 KB
  bf16_t* W2p = (bf16_t*)(ws + 589824);                    // 16*64*1KB = 1 MB
  float*  h2f = (float*)(ws + 589824 + 1048576);           // 8192*256*4 = 8 MB

  int n1 = K1T * NT * 512;
  pack_weights<<<(n1 + 255) / 256, 256, 0, stream>>>(W1, EMB + NH, n1, W1p);
  int n2 = K2T * NT * 512;
  pack_weights<<<(n2 + 255) / 256, 256, 0, stream>>>(W2, 2 * NH, n2, W2p);

  lstm_persist<<<BATCH / BT, 256, 0, stream>>>(features, embedding, b1, b2, W1p, W2p, h2f);
  dense_loss<<<BATCH / 8, 256, 0, stream>>>(h2f, Wd, bd, labels, out);
}
